// ScaleDotProductAttention_3135326126184
// MI455X (gfx1250) — matrix operand order repaired and checked by reading, not compiled, not run on hardware
//
#include <hip/hip_runtime.h>

typedef __attribute__((ext_vector_type(16))) __bf16 v16bf;
typedef __attribute__((ext_vector_type(8)))  __bf16 v8bf;
typedef __attribute__((ext_vector_type(4)))  __bf16 v4bf;
typedef __attribute__((ext_vector_type(8)))  float  v8f;
typedef __attribute__((ext_vector_type(4)))  float  v4f;

#define LDP 72   // padded LDS row stride in bf16 elements (144 B, 16B-aligned)

union V16U { v16bf v; v8bf h[2]; };

// reduce across the 16-lane half-wave (masks 1,2,4,8 stay inside each half)
__device__ __forceinline__ float hmax16(float v) {
  v = fmaxf(v, __shfl_xor(v, 1, 32));
  v = fmaxf(v, __shfl_xor(v, 2, 32));
  v = fmaxf(v, __shfl_xor(v, 4, 32));
  v = fmaxf(v, __shfl_xor(v, 8, 32));
  return v;
}
__device__ __forceinline__ float hsum16(float v) {
  v += __shfl_xor(v, 1, 32);
  v += __shfl_xor(v, 2, 32);
  v += __shfl_xor(v, 4, 32);
  v += __shfl_xor(v, 8, 32);
  return v;
}

__global__ __launch_bounds__(128)
void fa_fwd_kernel(const float* __restrict__ Q, const float* __restrict__ K,
                   const float* __restrict__ V, float* __restrict__ O)
{
  constexpr int L = 2048, D = 64;

  __shared__ __align__(16) __bf16 sK[64 * LDP];       // K block  [key][d]
  __shared__ __align__(16) __bf16 sV[64 * LDP];       // V block transposed [d][key]
  __shared__ __align__(16) __bf16 sP[4 * 32 * LDP];   // per-wave P tiles (2 x 16 rows)

  const int tid  = threadIdx.x;
  const int lane = tid & 31;
  const int wave = tid >> 5;
  const int n    = lane & 15;   // N index inside a 16-wide tile
  const int hi   = lane >> 4;   // half-wave select

  const int bh    = blockIdx.x >> 4;        // (b*H + h); 2048/128 = 16 q tiles/head
  const int qt    = blockIdx.x & 15;
  const long base = (long)bh * L * D;
  const int qbase = qt * 128 + wave * 32;   // 32 query rows per wave (2 M-tiles)

  // ---- Q: two M-tiles x two A-layout bf16 K-fragments, 1/sqrt(D)=0.125 folded in ----
  const int aklo = hi ? 8 : 0;   // A layout: this half holds K klo..+7 and klo+16..+23
  v16bf aq[2][2];
  for (int m = 0; m < 2; ++m) {
    const float* qrow = Q + base + (long)(qbase + m * 16 + n) * D;
    for (int f = 0; f < 2; ++f) {
      const int d0 = f * 32 + aklo;
      for (int j = 0; j < 8; ++j) {
        aq[m][f][j]     = (__bf16)(qrow[d0 + j]      * 0.125f);
        aq[m][f][j + 8] = (__bf16)(qrow[d0 + 16 + j] * 0.125f);
      }
    }
  }

  float mrow[2][8], lrow[2][8];
  v8f acc[2][4];
  for (int m = 0; m < 2; ++m)
    for (int r = 0; r < 8; ++r) { mrow[m][r] = -INFINITY; lrow[m][r] = 0.0f; }
  for (int m = 0; m < 2; ++m)
    for (int t = 0; t < 4; ++t) acc[m][t] = (v8f){};

  const int bklo = hi ? 16 : 0;  // unused: B loads use aklo, so B holds k aklo..+7 and aklo+16..+23 like A
  __bf16* pw = sP + wave * 32 * LDP;

  for (int kb = 0; kb < L / 64; ++kb) {
    const float* kblk = K + base + (long)kb * 64 * D;
    const float* vblk = V + base + (long)kb * 64 * D;

    __syncthreads();   // previous iteration done reading sK/sV
    // ---- stage K [key][d] and V^T [d][key] as bf16 (vectorized) ----
    for (int idx = tid; idx < 64 * 16; idx += 128) {
      const int ky = idx >> 4, d4 = (idx & 15) * 4;
      v4f kq = *(const v4f*)(kblk + ky * 64 + d4);
      v4f vq = *(const v4f*)(vblk + ky * 64 + d4);
      v4bf kh;
      for (int j = 0; j < 4; ++j) kh[j] = (__bf16)kq[j];
      *(v4bf*)(sK + ky * LDP + d4) = kh;
      for (int j = 0; j < 4; ++j) sV[(d4 + j) * LDP + ky] = (__bf16)vq[j];
    }
    if (kb + 1 < L / 64) {    // pull next block toward L2 (global_prefetch_b8)
      __builtin_prefetch(kblk + 64 * D + tid * 16, 0, 1);
      __builtin_prefetch(vblk + 64 * D + tid * 16, 0, 1);
    }
    __syncthreads();

    // ---- per M-tile: S = (Q/sqrt(D)) K^T, online softmax, P -> LDS ----
    for (int m = 0; m < 2; ++m) {
      v8f s[4];
      for (int t = 0; t < 4; ++t) {
        v8f c = (v8f){};
        for (int f = 0; f < 2; ++f) {
          V16U b;
          const __bf16* p = sK + (t * 16 + n) * LDP + f * 32 + aklo;
          b.h[0] = *(const v8bf*)(p);
          b.h[1] = *(const v8bf*)(p + 16);
          c = __builtin_amdgcn_wmma_f32_16x16x32_bf16(false, aq[m][f], false, b.v,
                                                      (short)0, c, false, false);
        }
        s[t] = c;
      }

      float alpha[8], rs[8];
      for (int r = 0; r < 8; ++r) {
        float rm = fmaxf(fmaxf(s[0][r], s[1][r]), fmaxf(s[2][r], s[3][r]));
        rm = hmax16(rm);
        const float nm = fmaxf(mrow[m][r], rm);
        alpha[r] = __expf(mrow[m][r] - nm);   // first pass: exp(-inf) = 0
        mrow[m][r] = nm;
      }
      for (int r = 0; r < 8; ++r) {
        float p0 = __expf(s[0][r] - mrow[m][r]);
        float p1 = __expf(s[1][r] - mrow[m][r]);
        float p2 = __expf(s[2][r] - mrow[m][r]);
        float p3 = __expf(s[3][r] - mrow[m][r]);
        s[0][r] = p0; s[1][r] = p1; s[2][r] = p2; s[3][r] = p3;
        rs[r] = hsum16(p0 + p1 + p2 + p3);
      }
      for (int r = 0; r < 8; ++r) lrow[m][r] = lrow[m][r] * alpha[r] + rs[r];
      for (int t = 0; t < 4; ++t)
        for (int r = 0; r < 8; ++r) acc[m][t][r] *= alpha[r];

      __bf16* pwm = pw + m * 16 * LDP;
      for (int t = 0; t < 4; ++t)
        for (int r = 0; r < 8; ++r)
          pwm[(r + 8 * hi) * LDP + t * 16 + n] = (__bf16)s[t][r];
    }
    asm volatile("s_wait_dscnt 0" ::: "memory");   // LDS per-wave in order; belt & braces

    // ---- reload P as A-layout fragments for both M-tiles ----
    v16bf ap[2][2];
    for (int m = 0; m < 2; ++m)
      for (int f = 0; f < 2; ++f) {
        V16U a;
        const __bf16* p = pw + (m * 16 + n) * LDP + f * 32 + aklo;
        a.h[0] = *(const v8bf*)(p);
        a.h[1] = *(const v8bf*)(p + 16);
        ap[m][f] = a.v;
      }

    // ---- O = O*alpha + P V ; V fragments shared across both M-tiles ----
    for (int t = 0; t < 4; ++t) {
      v8f c0 = acc[0][t], c1 = acc[1][t];
      for (int f = 0; f < 2; ++f) {
        V16U b;
        const __bf16* p = sV + (t * 16 + n) * LDP + f * 32 + aklo;
        b.h[0] = *(const v8bf*)(p);
        b.h[1] = *(const v8bf*)(p + 16);
        c0 = __builtin_amdgcn_wmma_f32_16x16x32_bf16(false, ap[0][f], false, b.v,
                                                     (short)0, c0, false, false);
        c1 = __builtin_amdgcn_wmma_f32_16x16x32_bf16(false, ap[1][f], false, b.v,
                                                     (short)0, c1, false, false);
      }
      acc[0][t] = c0; acc[1][t] = c1;
    }
  }

  // ---- normalize and store (C layout -> row-major global) ----
  for (int m = 0; m < 2; ++m)
    for (int r = 0; r < 8; ++r) {
      const float inv = 1.0f / lrow[m][r];
      const int qrow = qbase + m * 16 + r + 8 * hi;
      float* orow = O + base + (long)qrow * D;
      for (int t = 0; t < 4; ++t)
        orow[t * 16 + n] = acc[m][t][r] * inv;
    }
}

extern "C" void kernel_launch(void* const* d_in, const int* in_sizes, int n_in,
                              void* d_out, int out_size, void* d_ws, size_t ws_size,
                              hipStream_t stream) {
  const float* q = (const float*)d_in[0];
  const float* k = (const float*)d_in[1];
  const float* v = (const float*)d_in[2];
  float* o = (float*)d_out;
  constexpr int B = 4, H = 16, L = 2048;
  dim3 grid(B * H * (L / 128));
  fa_fwd_kernel<<<grid, 128, 0, stream>>>(q, k, v, o);
}
